// FormulaNet_4423816315426
// MI455X (gfx1250) — matrix operand order repaired and checked by reading, not compiled, not run on hardware
//
#include <hip/hip_runtime.h>
#include <stdint.h>

// ---------------- problem constants ----------------
#define D        256
#define NNODES   102400
#define NEDGES   204800
#define NTREE    204800
#define NGRAPH   64
#define BATCH    32
#define EPSBN    1e-5f

typedef __attribute__((ext_vector_type(16))) _Float16 v16h;
typedef __attribute__((ext_vector_type(8)))  float    v8f;
typedef __attribute__((ext_vector_type(4)))  unsigned uint4v;

union HFrag { v16h v; uint32_t u[8]; _Float16 h[16]; };

#define GEMM_PLAIN 0   // A (f32) as-is
#define GEMM_ADD   1   // A + A2 (x + d for FP block)
#define GEMM_AFF   2   // relu(A*affA[k] + affB[k])  (BN1+ReLU fused into W2 GEMM)
#define GEMM_F16A  3   // A already f16 (xh): async-copy tile straight to LDS

// ============================================================
// Weight pre-conversion: fp32 row-major W[256][256] -> f16
// B-fragment image: out[((slice*16+ct)*32+lane)*16 + e]
//   = W[slice*32 + (lane>>4)*8 + (e&7) + (e>>3)*16][ct*16 + (lane&15)]
// so a wave's B fragment is 2 contiguous b128 loads per lane.
// ============================================================
__global__ void wconv_k(const float* __restrict__ W, _Float16* __restrict__ out) {
    int t = blockIdx.x * 256 + threadIdx.x;     // 8192 threads, 8 halfs each
    #pragma unroll
    for (int i = 0; i < 8; ++i) {
        int f = t * 8 + i;
        int e = f & 15, lane = (f >> 4) & 31, ct = (f >> 9) & 15, slice = f >> 13;
        int k   = slice * 32 + (lane >> 4) * 8 + (e & 7) + (e >> 3) * 16;
        int col = ct * 16 + (lane & 15);
        out[f] = (_Float16)W[k * D + col];
    }
}

// ============================================================
// Workhorse: C[M x 256] = op(A)[M x 256] @ W[256 x 256] (+bias)
// Block = 256 threads = 8 waves; tile = 64 rows x 256 cols.
// A staged in LDS as f16 (async b128 copies for GEMM_F16A, VALU
// convert for fused modes); B fragments loaded directly from the
// pre-converted fragment image in global (L2 resident).
// ============================================================
__global__ __launch_bounds__(256) void gemm256_wmma(
    const float* __restrict__ A, const float* __restrict__ A2,
    const _Float16* __restrict__ Ah, const _Float16* __restrict__ Wf,
    const float* __restrict__ bias,
    const float* __restrict__ affA, const float* __restrict__ affB,
    float* __restrict__ C, int mode)
{
    __shared__ _Float16 Als[64 * 32];     // 64 rows x 32 K (f16), 4KB

    const int tid  = threadIdx.x;
    const int lane = tid & 31;
    const int w    = tid >> 5;
    const long r0  = (long)blockIdx.x * 64;
    const int rt   = w >> 1;               // row tile 0..3
    const int cg   = w & 1;                // col group 0..1 (8 col tiles each)

    const int m16    = lane & 15;
    const int hiHalf = lane >> 4;
    const int kbA    = hiHalf * 8;         // A frag K base (ISA 16-bit A layout)

    v8f acc[8] = {};

    for (int ks = 0; ks < 8; ++ks) {
        const int k0 = ks * 32;
        if (mode == GEMM_F16A) {
            // one async b128 per thread: 256 x 16B == 64 rows x 64B tile
            unsigned ldsa = (unsigned)(uintptr_t)&Als[(tid >> 2) * 32 + (tid & 3) * 8];
            unsigned long long gsrc = (unsigned long long)(uintptr_t)
                (Ah + (r0 + (tid >> 2)) * D + k0 + (tid & 3) * 8);
            asm volatile("global_load_async_to_lds_b128 %0, %1, off"
                         :: "v"(ldsa), "v"(gsrc) : "memory");
            asm volatile("s_wait_asynccnt 0x0" ::: "memory");
        } else {
            #pragma unroll
            for (int i = 0; i < 8; ++i) {
                int e = tid + i * 256;
                int row = e >> 5, kk = e & 31;
                int k = k0 + kk;
                float v = A[(r0 + row) * D + k];
                if (mode == GEMM_ADD)      v += A2[(r0 + row) * D + k];
                else if (mode == GEMM_AFF) v = fmaxf(v * affA[k] + affB[k], 0.f);
                Als[row * 32 + kk] = (_Float16)v;
            }
            if (ks < 7)   // global_prefetch_b8 of next K-slice of A
                __builtin_prefetch(&A[(r0 + (tid >> 2)) * D + k0 + 32 + (tid & 3) * 8], 0, 1);
        }
        __syncthreads();

        // A fragment (16x32): lanes 0-15 -> K 0..7 & 16..23; lanes 16-31 -> K 8..15 & 24..31
        HFrag af;
        {
            const uint32_t* p0 = (const uint32_t*)&Als[(rt * 16 + m16) * 32 + kbA];
            const uint32_t* p1 = (const uint32_t*)&Als[(rt * 16 + m16) * 32 + kbA + 16];
            #pragma unroll
            for (int j = 0; j < 4; ++j) { af.u[j] = p0[j]; af.u[4 + j] = p1[j]; }
        }
        // 8 col tiles: B fragment = 2 x b128 straight from fragment image
        #pragma unroll
        for (int j2 = 0; j2 < 8; ++j2) {
            int ct = cg * 8 + j2;
            HFrag bf;
            const uint4v* wf = (const uint4v*)(Wf + (((ks * 16 + ct) * 32 + lane) << 4));
            *(uint4v*)&bf.u[0] = wf[0];
            *(uint4v*)&bf.u[4] = wf[1];
            acc[j2] = __builtin_amdgcn_wmma_f32_16x16x32_f16(
                false, af.v, false, bf.v, (short)0, acc[j2], false, false);
        }
        __syncthreads();
    }

    // writeback: C/D layout: VGPR r -> M = rt*16 + r + hiHalf*8, N = m16
    #pragma unroll
    for (int j2 = 0; j2 < 8; ++j2) {
        int col = (cg * 8 + j2) * 16 + m16;
        float badd = bias ? bias[col] : 0.f;
        #pragma unroll
        for (int r = 0; r < 8; ++r) {
            int row = rt * 16 + r + hiHalf * 8;
            C[(r0 + row) * D + col] = acc[j2][r] + badd;
        }
    }
}

// ============================================================
// Support kernels (column-coalesced: thread = feature c)
// ============================================================
__global__ void embed_gather_k(const float* __restrict__ emb, const int* __restrict__ lab,
                               float* __restrict__ x, _Float16* __restrict__ xh,
                               int rpb, int M) {
    int c = threadIdx.x; long rb = (long)blockIdx.x * rpb;
    for (int i = 0; i < rpb; ++i) {
        long r = rb + i; if (r >= M) break;
        float v = emb[(long)lab[r] * D + c];
        x[r * D + c] = v; xh[r * D + c] = (_Float16)v;
    }
}

__global__ void gather2_stats_k(const float* __restrict__ pa, const float* __restrict__ pb,
                                const int* __restrict__ ed, float* __restrict__ h,
                                float* __restrict__ sums, float* __restrict__ sumsq,
                                int rpb, int M) {
    int c = threadIdx.x; long rb = (long)blockIdx.x * rpb;
    float s = 0.f, s2 = 0.f;
    for (int i = 0; i < rpb; ++i) {
        long r = rb + i; if (r >= M) break;
        int i0 = ed[r * 2], i1 = ed[r * 2 + 1];
        float v = pa[(long)i0 * D + c] + pb[(long)i1 * D + c];
        h[r * D + c] = v; s += v; s2 += v * v;
    }
    atomicAdd(&sums[c], s); atomicAdd(&sumsq[c], s2);
}

__global__ void gather3_stats_k(const float* __restrict__ pa, const float* __restrict__ pb,
                                const float* __restrict__ pc, const int* __restrict__ tr,
                                float* __restrict__ h, float* __restrict__ sums,
                                float* __restrict__ sumsq, int rpb, int M) {
    int c = threadIdx.x; long rb = (long)blockIdx.x * rpb;
    float s = 0.f, s2 = 0.f;
    for (int i = 0; i < rpb; ++i) {
        long r = rb + i; if (r >= M) break;
        int i0 = tr[r * 3], i1 = tr[r * 3 + 1], i2 = tr[r * 3 + 2];
        float v = pa[(long)i0 * D + c] + pb[(long)i1 * D + c] + pc[(long)i2 * D + c];
        h[r * D + c] = v; s += v; s2 += v * v;
    }
    atomicAdd(&sums[c], s); atomicAdd(&sumsq[c], s2);
}

__global__ void stats_cols_k(const float* __restrict__ t, float* __restrict__ sums,
                             float* __restrict__ sumsq, int rpb, int M) {
    int c = threadIdx.x; long rb = (long)blockIdx.x * rpb;
    float s = 0.f, s2 = 0.f;
    for (int i = 0; i < rpb; ++i) {
        long r = rb + i; if (r >= M) break;
        float v = t[r * D + c]; s += v; s2 += v * v;
    }
    atomicAdd(&sums[c], s); atomicAdd(&sumsq[c], s2);
}

__global__ void bn_finalize_k(const float* __restrict__ sums, const float* __restrict__ sumsq,
                              const float* __restrict__ g, const float* __restrict__ be,
                              float invM, float* __restrict__ affa, float* __restrict__ affb) {
    int c = threadIdx.x;
    float m = sums[c] * invM;
    float v = sumsq[c] * invM - m * m;
    float a = g[c] * rsqrtf(fmaxf(v, 0.f) + EPSBN);
    affa[c] = a; affb[c] = be[c] - m * a;
}

// d[idx[r]] += relu(t[r]*a + b)   (BN2+ReLU fused into segment_sum)
__global__ void scatter_bnrelu_k(const float* __restrict__ t, const int* __restrict__ idx,
                                 int stride, int which, const float* __restrict__ affa,
                                 const float* __restrict__ affb, float* __restrict__ dacc,
                                 int rpb, int M) {
    int c = threadIdx.x; float a = affa[c], b = affb[c];
    long rb = (long)blockIdx.x * rpb;
    for (int i = 0; i < rpb; ++i) {
        long r = rb + i; if (r >= M) break;
        int node = idx[r * stride + which];
        float v = fmaxf(t[r * D + c] * a + b, 0.f);
        atomicAdd(&dacc[(long)node * D + c], v);
    }
}

__global__ void bnrelu_write_k(const float* __restrict__ t, const float* __restrict__ affa,
                               const float* __restrict__ affb, float* __restrict__ xo,
                               _Float16* __restrict__ xh, int rpb, int M) {
    int c = threadIdx.x; float a = affa[c], b = affb[c];
    long rb = (long)blockIdx.x * rpb;
    for (int i = 0; i < rpb; ++i) {
        long r = rb + i; if (r >= M) break;
        float v = fmaxf(t[r * D + c] * a + b, 0.f);
        xo[r * D + c] = v; xh[r * D + c] = (_Float16)v;
    }
}

__global__ void segmax_k(const float* __restrict__ x, const int* __restrict__ ranges,
                         float* __restrict__ ge) {
    int g = blockIdx.x, c = threadIdx.x;
    long s = ranges[g * 2], e = ranges[g * 2 + 1];
    float m = -3.4e38f;
    for (long r = s; r < e; ++r) m = fmaxf(m, x[r * D + c]);
    ge[(long)g * D + c] = m;
}

__global__ void collect_gather_k(const float* __restrict__ ge, const int* __restrict__ conj,
                                 const int* __restrict__ stmt, float* __restrict__ hcat, int t) {
    int b = blockIdx.x, c = threadIdx.x;
    long row = (long)t * BATCH + b;
    hcat[row * 512 + c]       = ge[(long)conj[b] * D + c];
    hcat[row * 512 + 256 + c] = ge[(long)stmt[b] * D + c];
}

__global__ void cls_gemm1_k(const float* __restrict__ hcat, const float* __restrict__ W1,
                            const float* __restrict__ b1, float* __restrict__ h1,
                            float* __restrict__ sums, float* __restrict__ sumsq) {
    int row = blockIdx.x, c = threadIdx.x;
    float acc = b1[c];
    for (int k = 0; k < 512; ++k) acc += hcat[(long)row * 512 + k] * W1[(long)k * D + c];
    h1[(long)row * D + c] = acc;
    atomicAdd(&sums[c], acc); atomicAdd(&sumsq[c], acc * acc);
}

__global__ void cls_out_k(const float* __restrict__ h1, const float* __restrict__ affa,
                          const float* __restrict__ affb, const float* __restrict__ W2,
                          const float* __restrict__ b2, float* __restrict__ out) {
    __shared__ float red0[256], red1[256];
    int row = blockIdx.x, k = threadIdx.x;
    float hv = fmaxf(h1[(long)row * D + k] * affa[k] + affb[k], 0.f);
    red0[k] = hv * W2[k * 2]; red1[k] = hv * W2[k * 2 + 1];
    __syncthreads();
    for (int s = 128; s > 0; s >>= 1) {
        if (k < s) { red0[k] += red0[k + s]; red1[k] += red1[k + s]; }
        __syncthreads();
    }
    if (k == 0) { out[row * 2] = red0[0] + b2[0]; out[row * 2 + 1] = red1[0] + b2[1]; }
}

// ============================================================
// Host orchestration
// ============================================================
struct B2p { const float *W1a,*W1b,*W2,*b1b,*b2,*be1,*be2,*g1,*g2; };
struct B3p { const float *W1a,*W1b,*W1c,*W2,*b1c,*b2,*be1,*be2,*g1,*g2; };
struct FPp { const float *W,*b,*be,*g; };
struct StepP { B3p FH; B2p FI; B3p FL; B2p FO; FPp FP; B3p FR; };

extern "C" void kernel_launch(void* const* d_in, const int* in_sizes, int n_in,
                              void* d_out, int out_size, void* d_ws, size_t ws_size,
                              hipStream_t stream) {
    (void)in_sizes; (void)n_in; (void)out_size; (void)ws_size;

    // ---- input mapping: jax pytree leaf order (dict keys sorted at each level) ----
    const int* conj_idx = (const int*)d_in[0];
    const int* edges    = (const int*)d_in[1];
    const int* labels   = (const int*)d_in[2];
    const int* ranges   = (const int*)d_in[3];
    const float* clsW1 = (const float*)d_in[4];
    const float* clsW2 = (const float*)d_in[5];
    const float* clsb1 = (const float*)d_in[6];
    const float* clsb2 = (const float*)d_in[7];
    const float* clsbe = (const float*)d_in[8];
    const float* clsg  = (const float*)d_in[9];
    const float* embed = (const float*)d_in[10];
    int p = 11;
    auto rd = [&]() -> const float* { return (const float*)d_in[p++]; };
    auto rdB2 = [&](B2p& q){ q.W1a=rd(); q.W1b=rd(); q.W2=rd(); q.b1b=rd(); q.b2=rd();
                             q.be1=rd(); q.be2=rd(); q.g1=rd(); q.g2=rd(); };
    auto rdB3 = [&](B3p& q){ q.W1a=rd(); q.W1b=rd(); q.W1c=rd(); q.W2=rd(); q.b1c=rd();
                             q.b2=rd(); q.be1=rd(); q.be2=rd(); q.g1=rd(); q.g2=rd(); };
    StepP steps[2];
    for (int s = 0; s < 2; ++s) {
        rdB3(steps[s].FH); rdB2(steps[s].FI); rdB3(steps[s].FL); rdB2(steps[s].FO);
        steps[s].FP.W=rd(); steps[s].FP.b=rd(); steps[s].FP.be=rd(); steps[s].FP.g=rd();
        rdB3(steps[s].FR);
    }
    const int* stmt_idx = (const int*)d_in[p++];
    const int* tre      = (const int*)d_in[p++];
    float* out = (float*)d_out;

    // ---- workspace carve (16-float aligned) ----
    float* ws = (float*)d_ws;
    size_t off = 0;
    auto take = [&](size_t n){ float* q = ws + off; off += (n + 15) & ~(size_t)15; return q; };
    float* x    = take((size_t)NNODES * D);
    float* dacc = take((size_t)NNODES * D);
    float* pA   = take((size_t)NNODES * D);
    float* pB   = take((size_t)NNODES * D);
    float* pC   = take((size_t)NNODES * D);
    float* hbuf = take((size_t)NEDGES * D);            // edge/treelet activations (in-place -> t)
    _Float16* xh = (_Float16*)take((size_t)NNODES * D / 2);   // f16 mirror of x
    _Float16* wfragBase = (_Float16*)take((size_t)38 * 65536 / 2); // 38 weight images
    float* ge   = take((size_t)NGRAPH * D);
    float* hcat = take((size_t)96 * 512);
    float* h1   = take((size_t)96 * D);
    float* stat = take(1024);                          // sums|sumsq|affa|affb
    float* sums = stat, *sumsq = stat + 256, *affa = stat + 512, *affb = stat + 768;

    // ---- pre-convert every 256x256 weight to the B-fragment image ----
    int wcount = 0;
    auto conv = [&](const float* W) -> const _Float16* {
        _Float16* dst = wfragBase + (size_t)wcount * 65536; ++wcount;
        wconv_k<<<32, 256, 0, stream>>>(W, dst);
        return dst;
    };
    struct B2f { const _Float16 *W1a,*W1b,*W2; };
    struct B3f { const _Float16 *W1a,*W1b,*W1c,*W2; };
    B2f fiF[2], foF[2]; B3f flF[2], fhF[2], frF[2]; const _Float16* fpF[2];
    for (int s = 0; s < 2; ++s) {
        const StepP& sp = steps[s];
        fiF[s] = { conv(sp.FI.W1a), conv(sp.FI.W1b), conv(sp.FI.W2) };
        foF[s] = { conv(sp.FO.W1a), conv(sp.FO.W1b), conv(sp.FO.W2) };
        flF[s] = { conv(sp.FL.W1a), conv(sp.FL.W1b), conv(sp.FL.W1c), conv(sp.FL.W2) };
        fhF[s] = { conv(sp.FH.W1a), conv(sp.FH.W1b), conv(sp.FH.W1c), conv(sp.FH.W2) };
        frF[s] = { conv(sp.FR.W1a), conv(sp.FR.W1b), conv(sp.FR.W1c), conv(sp.FR.W2) };
        fpF[s] = conv(sp.FP.W);
    }

    const int RPB = 256;
    const int GB_E = NEDGES / RPB, GB_N = NNODES / RPB;

    auto resetStats = [&]{ hipMemsetAsync(stat, 0, 512 * sizeof(float), stream); };
    auto gemm = [&](const float* A, const float* A2, const _Float16* Ah, const _Float16* Wf,
                    const float* bias, const float* aa, const float* ab,
                    float* C, int M, int mode){
        gemm256_wmma<<<M / 64, 256, 0, stream>>>(A, A2, Ah, Wf, bias, aa, ab, C, mode);
    };
    auto collect = [&](int t){
        segmax_k<<<NGRAPH, 256, 0, stream>>>(x, ranges, ge);
        collect_gather_k<<<BATCH, 256, 0, stream>>>(ge, conj_idx, stmt_idx, hcat, t);
    };

    embed_gather_k<<<GB_N, 256, 0, stream>>>(embed, labels, x, xh, RPB, NNODES);
    collect(0);

    for (int s = 0; s < 2; ++s) {
        const StepP& sp = steps[s];
        hipMemsetAsync(dacc, 0, (size_t)NNODES * D * sizeof(float), stream);

        // ---- edge blocks FI (scatter @ e1), FO (scatter @ e0) ----
        const B2p* eb[2] = { &sp.FI, &sp.FO };
        const B2f* ef[2] = { &fiF[s], &foF[s] };
        const int  ew[2] = { 1, 0 };
        for (int q = 0; q < 2; ++q) {
            const B2p& b = *eb[q]; const B2f& f = *ef[q];
            gemm(nullptr, nullptr, xh, f.W1a, nullptr, nullptr, nullptr, pA, NNODES, GEMM_F16A);
            gemm(nullptr, nullptr, xh, f.W1b, b.b1b,  nullptr, nullptr, pB, NNODES, GEMM_F16A);
            resetStats();
            gather2_stats_k<<<GB_E, 256, 0, stream>>>(pA, pB, edges, hbuf, sums, sumsq, RPB, NEDGES);
            bn_finalize_k<<<1, 256, 0, stream>>>(sums, sumsq, b.g1, b.be1, 1.f / NEDGES, affa, affb);
            resetStats();
            gemm(hbuf, nullptr, nullptr, f.W2, b.b2, affa, affb, hbuf, NEDGES, GEMM_AFF); // in-place
            stats_cols_k<<<GB_E, 256, 0, stream>>>(hbuf, sums, sumsq, RPB, NEDGES);
            bn_finalize_k<<<1, 256, 0, stream>>>(sums, sumsq, b.g2, b.be2, 1.f / NEDGES, affa, affb);
            scatter_bnrelu_k<<<GB_E, 256, 0, stream>>>(hbuf, edges, 2, ew[q], affa, affb, dacc, RPB, NEDGES);
        }

        // ---- treelet blocks FL (@t1), FH (@t0), FR (@t2) ----
        const B3p* tb[3] = { &sp.FL, &sp.FH, &sp.FR };
        const B3f* tf[3] = { &flF[s], &fhF[s], &frF[s] };
        const int  tw[3] = { 1, 0, 2 };
        for (int q = 0; q < 3; ++q) {
            const B3p& b = *tb[q]; const B3f& f = *tf[q];
            gemm(nullptr, nullptr, xh, f.W1a, nullptr, nullptr, nullptr, pA, NNODES, GEMM_F16A);
            gemm(nullptr, nullptr, xh, f.W1b, nullptr, nullptr, nullptr, pB, NNODES, GEMM_F16A);
            gemm(nullptr, nullptr, xh, f.W1c, b.b1c,  nullptr, nullptr, pC, NNODES, GEMM_F16A);
            resetStats();
            gather3_stats_k<<<GB_E, 256, 0, stream>>>(pA, pB, pC, tre, hbuf, sums, sumsq, RPB, NTREE);
            bn_finalize_k<<<1, 256, 0, stream>>>(sums, sumsq, b.g1, b.be1, 1.f / NTREE, affa, affb);
            resetStats();
            gemm(hbuf, nullptr, nullptr, f.W2, b.b2, affa, affb, hbuf, NTREE, GEMM_AFF);  // in-place
            stats_cols_k<<<GB_E, 256, 0, stream>>>(hbuf, sums, sumsq, RPB, NTREE);
            bn_finalize_k<<<1, 256, 0, stream>>>(sums, sumsq, b.g2, b.be2, 1.f / NTREE, affa, affb);
            scatter_bnrelu_k<<<GB_E, 256, 0, stream>>>(hbuf, tre, 3, tw[q], affa, affb, dacc, RPB, NTREE);
        }

        // ---- FP: x = relu(bn((x + d) @ W + b)) ----
        resetStats();
        gemm(x, dacc, nullptr, fpF[s], sp.FP.b, nullptr, nullptr, pA, NNODES, GEMM_ADD);
        stats_cols_k<<<GB_N, 256, 0, stream>>>(pA, sums, sumsq, RPB, NNODES);
        bn_finalize_k<<<1, 256, 0, stream>>>(sums, sumsq, sp.FP.g, sp.FP.be, 1.f / NNODES, affa, affb);
        bnrelu_write_k<<<GB_N, 256, 0, stream>>>(pA, affa, affb, x, xh, RPB, NNODES);

        collect(s + 1);
    }

    // ---- classifier: out = relu(bn(hcat @ W1 + b1)) @ W2 + b2 ----
    resetStats();
    cls_gemm1_k<<<96, 256, 0, stream>>>(hcat, clsW1, clsb1, h1, sums, sumsq);
    bn_finalize_k<<<1, 256, 0, stream>>>(sums, sumsq, clsg, clsbe, 1.f / 96.f, affa, affb);
    cls_out_k<<<96, 256, 0, stream>>>(h1, affa, affb, clsW2, clsb2, out);
}
